// DeformableCrossAttention_7413113553235
// MI455X (gfx1250) — matrix operand order repaired and checked by reading, not compiled, not run on hardware
//
#include <hip/hip_runtime.h>
#include <hip/hip_bf16.h>
#include <math.h>

// ---------------------------------------------------------------------------
// Deformable cross attention forward for MI455X (gfx1250, wave32, WMMA).
// All matmuls run through a bf16->f32 WMMA GEMM (v_wmma_f32_16x16x32_bf16).
// Weights are padded to N%128==0 at conversion so the GEMM hot loop has no
// ragged-edge branches; each wave computes a 32x32 slab (4 WMMAs / k-tile).
// LDS tiles are typed `unsigned`; B-tile transpose packing is done entirely
// with 32-bit integer ops (no __bf16 scalar extraction -- avoids a clang
// fold that collapsed per-element bit_casts of __bf16 vector lanes).
// ---------------------------------------------------------------------------

typedef __bf16 bf16;
typedef __attribute__((ext_vector_type(16))) __bf16 v16bf;
typedef __attribute__((ext_vector_type(8)))  __bf16 v8bf;
typedef __attribute__((ext_vector_type(8)))  float  v8f;
typedef __attribute__((ext_vector_type(4)))  unsigned u32x4;

static constexpr int Bn    = 2;      // batch
static constexpr int Cn    = 256;    // channels
static constexpr int H0    = 80, W0 = 80;
static constexpr int N0    = H0 * W0;          // 6400
static constexpr int HEADS = 8;
static constexpr int HDd   = 32;               // head dim
static constexpr int K9    = Cn * 9;           // 2304 (im2col K)

__device__ __forceinline__ bf16 f2bf(float f) {
  unsigned u = __builtin_bit_cast(unsigned, f);
  u += 0x7fffu + ((u >> 16) & 1u);             // round-to-nearest-even
  return __builtin_bit_cast(bf16, (unsigned short)(u >> 16));
}
__device__ __forceinline__ bf16 bf0() {
  return __builtin_bit_cast(bf16, (unsigned short)0);
}

// Build a 16-element bf16 fragment from two 16B LDS slots (aliasing-safe).
__device__ __forceinline__ v16bf frag16(const unsigned* p0, const unsigned* p1) {
  v8bf lo, hi;
  __builtin_memcpy(&lo, p0, 16);
  __builtin_memcpy(&hi, p1, 16);
  return __builtin_shufflevector(lo, hi,
      0,1,2,3,4,5,6,7,8,9,10,11,12,13,14,15);
}

// ---------------------------------------------------------------------------
// Tiled bf16 WMMA GEMM:  C[M,N] = act(A[M,K] @ B[K,ldb] + bias)
// A row-major (lda=K), B row-major with padded ldb (ldb%128==0, zero-padded
// cols >= N).  Block tile 64x128, 8 waves in 2x4 grid, 32x32 slab per wave.
// K%32==0.  Cf (f32) / Cb (bf16) outputs, either may be null.
//
// LDS layout (unsigned words, 2 bf16 per word along K):
//   sA[m][kw]  : A tile, kw = K/2 word index (row = 20 words = 80B)
//   sB[n][kw]  : transposed B tile, word packs {K=2kw (lo), K=2kw+1 (hi)}
// Fragment maps (ISA 7.12.2, 16-bit operands, wave32):
//   A: lane m = lane%16, K halves {kh..kh+7} and {16+kh..+7}, kh = 8*lane[4]
//   B: lane n = lane%16, K halves {kh..kh+7} and {16+kh..+7}, kh = 8*lane[4]
// ---------------------------------------------------------------------------
__global__ __launch_bounds__(256)
void k_gemm_bf16(const bf16* __restrict__ A, const bf16* __restrict__ Bm,
                 const float* __restrict__ bias,
                 float* __restrict__ Cf, bf16* __restrict__ Cb,
                 int M, int N, int K, int ldb, int act)
{
  __shared__ __attribute__((aligned(16))) unsigned sA[64][20];
  __shared__ __attribute__((aligned(16))) unsigned sB[128][20];

  const int tid  = threadIdx.x;
  const int lane = tid & 31;
  const int wave = tid >> 5;
  const int wm   = wave >> 2;          // 0..1  (M direction, 32 rows each)
  const int wn   = wave & 3;           // 0..3  (N direction, 32 cols each)

  const int m0 = blockIdx.y * 64;
  const int n0 = blockIdx.x * 128;

  const v8f z8 = {0.f,0.f,0.f,0.f,0.f,0.f,0.f,0.f};
  v8f acc00 = z8, acc01 = z8, acc10 = z8, acc11 = z8;

  // A staging: row la_r (0..63), 8 bf16 (= 4 words) at word col la_w
  const int  la_r = tid >> 2;
  const int  la_w = (tid & 3) * 4;
  const bool aIn  = (m0 + la_r) < M;
  const bf16* aPtr = A + (size_t)(m0 + la_r) * K + la_w * 2;

  // B staging: k-pair 2*kw (kw = tid>>4, 0..15), 8 cols at ng
  const int kw = tid >> 4;
  const int ng = (tid & 15) * 8;
  const bf16* bPtr = Bm + (size_t)(kw * 2) * ldb + (n0 + ng);
  const size_t bStep = (size_t)32 * ldb;

  for (int k0 = 0; k0 < K; k0 += 32) {
    // ---- stage A tile ----
    {
      v8bf av = {bf0(),bf0(),bf0(),bf0(),bf0(),bf0(),bf0(),bf0()};
      if (aIn) av = *(const v8bf*)(aPtr + k0);
      __builtin_memcpy(&sA[la_r][la_w], &av, 16);
    }
    // ---- stage B tile transposed, packed pairs; 32-bit ops only ----
    {
      v8bf b0 = *(const v8bf*)(bPtr);
      v8bf b1 = *(const v8bf*)(bPtr + ldb);
      bPtr += bStep;
      const u32x4 u0 = __builtin_bit_cast(u32x4, b0);
      const u32x4 u1 = __builtin_bit_cast(u32x4, b1);
#pragma unroll
      for (int j = 0; j < 4; ++j) {
        // n = ng+2j   gets {K=2kw: b0 elem 2j,   K=2kw+1: b1 elem 2j}
        // n = ng+2j+1 gets {K=2kw: b0 elem 2j+1, K=2kw+1: b1 elem 2j+1}
        sB[ng + 2*j    ][kw] = (u1[j] << 16) | (u0[j] & 0xffffu);
        sB[ng + 2*j + 1][kw] = (u1[j] & 0xffff0000u) | (u0[j] >> 16);
      }
    }
    __syncthreads();

    const int mr  = lane & 15;
    const int kh2 = (lane >> 4) * 4;   // A K-half in words: {0,4}
    const int kb2 = (lane >> 4) * 4;   // B K-half in words: {0,4}

    const int ar0 = wm * 32 + mr, ar1 = ar0 + 16;
    const int nb0 = wn * 32 + mr, nb1 = nb0 + 16;

    v16bf a0 = frag16(&sA[ar0][kh2], &sA[ar0][8 + kh2]);
    v16bf a1 = frag16(&sA[ar1][kh2], &sA[ar1][8 + kh2]);
    v16bf b0 = frag16(&sB[nb0][kb2], &sB[nb0][8 + kb2]);
    v16bf b1 = frag16(&sB[nb1][kb2], &sB[nb1][8 + kb2]);

    acc00 = __builtin_amdgcn_wmma_f32_16x16x32_bf16(
        false, a0, false, b0, (short)0, acc00, false, false);
    acc01 = __builtin_amdgcn_wmma_f32_16x16x32_bf16(
        false, a0, false, b1, (short)0, acc01, false, false);
    acc10 = __builtin_amdgcn_wmma_f32_16x16x32_bf16(
        false, a1, false, b0, (short)0, acc10, false, false);
    acc11 = __builtin_amdgcn_wmma_f32_16x16x32_bf16(
        false, a1, false, b1, (short)0, acc11, false, false);
    __syncthreads();
  }

  // ---- epilogue: bias + optional exact GELU + f32/bf16 stores ----
  const int cn0   = n0 + wn * 32 + (lane & 15);        // D: N = lane%16
  const int cn1   = cn0 + 16;
  const int rbase = m0 + wm * 32 + ((lane >> 4) * 8);  // D: M = r + 8*lane[4]
  float bv0 = 0.f, bv1 = 0.f;
  if (bias && cn0 < N) bv0 = bias[cn0];
  if (bias && cn1 < N) bv1 = bias[cn1];
#pragma unroll
  for (int r = 0; r < 8; ++r) {
#pragma unroll
    for (int s = 0; s < 2; ++s) {
      const int gm = rbase + s * 16 + r;
      float v0 = (s ? acc10[r] : acc00[r]) + bv0;
      float v1 = (s ? acc11[r] : acc01[r]) + bv1;
      if (act == 1) {
        v0 = 0.5f * v0 * (1.f + erff(v0 * 0.70710678118654752f));
        v1 = 0.5f * v1 * (1.f + erff(v1 * 0.70710678118654752f));
      }
      if (gm < M) {
        if (cn0 < N) {
          const size_t o = (size_t)gm * N + cn0;
          if (Cf) Cf[o] = v0;
          if (Cb) Cb[o] = f2bf(v0);
        }
        if (cn1 < N) {
          const size_t o = (size_t)gm * N + cn1;
          if (Cf) Cf[o] = v1;
          if (Cb) Cb[o] = f2bf(v1);
        }
      }
    }
  }
}

// ---------------------------------------------------------------------------
// Depthwise 3x3 pos-conv (pad 1) + residual; emits NCHW and seq layouts.
// ---------------------------------------------------------------------------
__global__ __launch_bounds__(256)
void k_dwconv_add(float* __restrict__ chw, float* __restrict__ seq,
                  const float* __restrict__ x, const float* __restrict__ w)
{
  const size_t total = (size_t)Bn * Cn * N0;
  const size_t idx = (size_t)blockIdx.x * 256 + threadIdx.x;
  if (idx >= total) return;
  const int wq = idx % W0;
  const int h  = (idx / W0) % H0;
  const int c  = (idx / N0) % Cn;
  const int b  = idx / ((size_t)N0 * Cn);
  const float* xp = x + ((size_t)(b * Cn + c)) * N0;
  const float* wp = w + c * 9;
  float s = xp[h * W0 + wq];
#pragma unroll
  for (int kh = 0; kh < 3; ++kh) {
    const int hi = h + kh - 1;
    if (hi < 0 || hi >= H0) continue;
#pragma unroll
    for (int kw = 0; kw < 3; ++kw) {
      const int wi = wq + kw - 1;
      if (wi < 0 || wi >= W0) continue;
      s += xp[hi * W0 + wi] * wp[kh * 3 + kw];
    }
  }
  chw[idx] = s;
  seq[((size_t)b * N0 + h * W0 + wq) * Cn + c] = s;
}

// im2col for 3x3 stride-2 pad-1 conv, fused f32->bf16.
__global__ __launch_bounds__(256)
void k_im2col(bf16* __restrict__ dst, const float* __restrict__ chw,
              int Hin, int Win, int Hout, int Wout)
{
  const size_t total = (size_t)Bn * Hout * Wout * K9;
  const size_t idx = (size_t)blockIdx.x * 256 + threadIdx.x;
  if (idx >= total) return;
  const int k    = idx % K9;
  const size_t row = idx / K9;
  const int pos  = row % (Hout * Wout);
  const int b    = row / (Hout * Wout);
  const int c  = k / 9, kk = k % 9, kh = kk / 3, kw = kk % 3;
  const int ho = pos / Wout, wo = pos % Wout;
  const int hi = ho * 2 + kh - 1, wi = wo * 2 + kw - 1;
  float v = 0.f;
  if (hi >= 0 && hi < Hin && wi >= 0 && wi < Win)
    v = chw[(((size_t)b * Cn + c) * Hin + hi) * Win + wi];
  dst[idx] = f2bf(v);
}

// OIHW conv weight -> row-major (K=C*9, N=C) bf16.
__global__ __launch_bounds__(256)
void k_conv_w_kn(bf16* __restrict__ dst, const float* __restrict__ w)
{
  const size_t total = (size_t)K9 * Cn;
  const size_t idx = (size_t)blockIdx.x * 256 + threadIdx.x;
  if (idx >= total) return;
  const int k = idx / Cn, o = idx % Cn;
  dst[idx] = f2bf(w[(size_t)o * K9 + k]);
}

// f32 (Kd x Nd) weight -> bf16 (Kd x Npad), zero-padded cols >= Nd.
__global__ __launch_bounds__(256)
void k_cvt_pad(bf16* __restrict__ dst, const float* __restrict__ src,
               int Kd, int Nd, int Npad)
{
  const size_t total = (size_t)Kd * Npad;
  const size_t idx = (size_t)blockIdx.x * 256 + threadIdx.x;
  if (idx >= total) return;
  const int k = idx / Npad, n = idx % Npad;
  dst[idx] = (n < Nd) ? f2bf(src[(size_t)k * Nd + n]) : bf0();
}

__global__ __launch_bounds__(256)
void k_cvt_bf16(bf16* __restrict__ dst, const float* __restrict__ src, size_t n)
{
  const size_t idx = (size_t)blockIdx.x * 256 + threadIdx.x;
  if (idx < n) dst[idx] = f2bf(src[idx]);
}

// y = seq + level_embed (broadcast over rows); emits f32 + bf16.
__global__ __launch_bounds__(256)
void k_add_embed(float* __restrict__ outf, bf16* __restrict__ outb,
                 const float* __restrict__ seq, const float* __restrict__ emb,
                 size_t total)
{
  const size_t idx = (size_t)blockIdx.x * 256 + threadIdx.x;
  if (idx >= total) return;
  const float v = seq[idx] + emb[idx & (Cn - 1)];
  outf[idx] = v;
  outb[idx] = f2bf(v);
}

// Wave-per-row layernorm over C=256 (optional residual, optional bf16 out).
__global__ __launch_bounds__(256)
void k_layernorm(float* __restrict__ outf, bf16* __restrict__ outb,
                 const float* __restrict__ x, const float* __restrict__ res,
                 const float* __restrict__ g, const float* __restrict__ bb,
                 int rows)
{
  const int row  = (int)(((size_t)blockIdx.x * 256 + threadIdx.x) >> 5);
  const int lane = threadIdx.x & 31;
  if (row >= rows) return;
  const float* xp = x + (size_t)row * Cn;
  const float* rp = res ? res + (size_t)row * Cn : nullptr;
  float v[8], s = 0.f, s2 = 0.f;
#pragma unroll
  for (int i = 0; i < 8; ++i) {
    const int c = i * 32 + lane;
    float t = xp[c];
    if (rp) t += rp[c];
    v[i] = t; s += t; s2 += t * t;
  }
#pragma unroll
  for (int m = 16; m >= 1; m >>= 1) {
    s  += __shfl_xor(s,  m, 32);
    s2 += __shfl_xor(s2, m, 32);
  }
  const float mu  = s * (1.f / Cn);
  const float var = s2 * (1.f / Cn) - mu * mu;
  const float rs  = rsqrtf(var + 1e-5f);
#pragma unroll
  for (int i = 0; i < 8; ++i) {
    const int c = i * 32 + lane;
    const float y = (v[i] - mu) * rs * g[c] + bb[c];
    outf[(size_t)row * Cn + c] = y;
    if (outb) outb[(size_t)row * Cn + c] = f2bf(y);
  }
}

// In-place softmax over groups of 24 (HEADS groups per 96-wide row).
__global__ __launch_bounds__(256)
void k_softmax24(float* __restrict__ aw, int total)
{
  const int t = blockIdx.x * 256 + threadIdx.x;
  if (t >= total) return;
  const int h = t % HEADS;
  float* p = aw + (size_t)(t / HEADS) * 96 + h * 24;
  float m = -1e30f;
#pragma unroll
  for (int i = 0; i < 24; ++i) m = fmaxf(m, p[i]);
  float e[24], s = 0.f;
#pragma unroll
  for (int i = 0; i < 24; ++i) { e[i] = __expf(p[i] - m); s += e[i]; }
  const float inv = 1.f / s;
#pragma unroll
  for (int i = 0; i < 24; ++i) p[i] = e[i] * inv;
}

__device__ __forceinline__ float bilin_corner(const float* v, int xi, int yi,
                                              int Wd, int Hd, float wt)
{
  if ((unsigned)xi < (unsigned)Wd && (unsigned)yi < (unsigned)Hd)
    return wt * v[(size_t)(yi * Wd + xi) * Cn];
  return 0.f;
}

// Wave-per-(b,n,head); 32 lanes = 32 head-dims. 3 levels x 4 pts bilinear.
__global__ __launch_bounds__(256)
void k_sample(float* __restrict__ out, const float* __restrict__ off,
              const float* __restrict__ aw,
              const float* __restrict__ v0, const float* __restrict__ v1,
              const float* __restrict__ v2)
{
  const int gw   = (int)(((size_t)blockIdx.x * 256 + threadIdx.x) >> 5);
  const int lane = threadIdx.x & 31;
  const int total = Bn * N0 * HEADS;
  if (gw >= total) return;
  const int h = gw % HEADS;
  const int n = (gw / HEADS) % N0;
  const int b = gw / (HEADS * N0);

  const float* vt[3] = { v0 + (size_t)b * N0  * Cn,
                         v1 + (size_t)b * 1600 * Cn,
                         v2 + (size_t)b * 400  * Cn };
  const int Wd[3] = {80, 40, 20};

  const float refx = ((n % W0) + 0.5f) * (1.f / W0);
  const float refy = ((n / W0) + 0.5f) * (1.f / H0);
  const size_t qi = (size_t)b * N0 + n;
  const float* offp = off + qi * 192 + h * 24;  // (h,l,p,2) -> h*24+l*8+p*2
  const float* awp  = aw  + qi * 96  + h * 24;  // (h,l,p)   -> h*24+l*4+p

  float accv = 0.f;
#pragma unroll
  for (int l = 0; l < 3; ++l) {
    const int W = Wd[l], H = Wd[l];
    const float* v = vt[l] + h * HDd + lane;
#pragma unroll
    for (int p = 0; p < 4; ++p) {
      const float ox = offp[l * 8 + p * 2 + 0];
      const float oy = offp[l * 8 + p * 2 + 1];
      const float x = refx * W + ox - 0.5f;
      const float y = refy * H + oy - 0.5f;
      const float xf = floorf(x), yf = floorf(y);
      const float wx = x - xf, wy = y - yf;
      const int x0 = (int)xf, y0 = (int)yf;
      const float a = awp[l * 4 + p];
      float s = bilin_corner(v, x0,     y0,     W, H, (1.f - wx) * (1.f - wy))
              + bilin_corner(v, x0 + 1, y0,     W, H, wx * (1.f - wy))
              + bilin_corner(v, x0,     y0 + 1, W, H, (1.f - wx) * wy)
              + bilin_corner(v, x0 + 1, y0 + 1, W, H, wx * wy);
      accv += a * s;
    }
  }
  out[qi * Cn + h * HDd + lane] = accv;
}

// seq (B,HW,C) -> chw (B,C,HW)
__global__ __launch_bounds__(256)
void k_seq_to_chw(float* __restrict__ chw, const float* __restrict__ seq, int HW)
{
  const size_t total = (size_t)Bn * Cn * HW;
  const size_t idx = (size_t)blockIdx.x * 256 + threadIdx.x;
  if (idx >= total) return;
  const int pos = idx % HW;
  const int c   = (idx / HW) % Cn;
  const int b   = idx / ((size_t)HW * Cn);
  chw[idx] = seq[((size_t)b * HW + pos) * Cn + c];
}

// ---------------------------------------------------------------------------
// Host orchestration
// ---------------------------------------------------------------------------
namespace {
struct BranchP {
  const float *ds_b0, *ds_b1, *conv0, *conv1, *g0, *g1;
  const float *W1, *W2, *b1, *b2, *embed;
  const float *n1_b, *n1_g, *n2_b, *n2_g, *pos, *feats;
};
struct AttnP {
  const float *W_attn, *W_off, *W_out, *W_val;
  const float *b_attn, *b_off, *b_out, *b_val;
};
}

extern "C" void kernel_launch(void* const* d_in, const int* in_sizes, int n_in,
                              void* d_out, int out_size, void* d_ws, size_t ws_size,
                              hipStream_t stream) {
  (void)in_sizes; (void)n_in; (void)out_size; (void)ws_size;
  auto IN = [&](int i) { return (const float*)d_in[i]; };

  // params flattened in jax tree order (dict keys sorted, lists in order):
  // 0 point_feats, 1 line_feats,
  // 2..9   l2p {W_attn,W_off,W_out,W_val,b_attn,b_off,b_out,b_val}
  // 10..25 line {ds_b0,ds_b1,ds_conv0,ds_conv1,ds_g0,ds_g1, ffn{W1,W2,b1,b2},
  //              level_embed, n1_b,n1_g,n2_b,n2_g, pos}
  // 26..33 p2l, 34..49 point (same layout as line)
  const AttnP l2p = { IN(2), IN(3), IN(4), IN(5), IN(6), IN(7), IN(8), IN(9) };
  const AttnP p2l = { IN(26),IN(27),IN(28),IN(29),IN(30),IN(31),IN(32),IN(33) };
  const BranchP line  = { IN(10),IN(11),IN(12),IN(13),IN(14),IN(15),
                          IN(16),IN(17),IN(18),IN(19),IN(20),
                          IN(21),IN(22),IN(23),IN(24),IN(25), IN(1) };
  const BranchP point = { IN(34),IN(35),IN(36),IN(37),IN(38),IN(39),
                          IN(40),IN(41),IN(42),IN(43),IN(44),
                          IN(45),IN(46),IN(47),IN(48),IN(49), IN(0) };
  const BranchP brp[2] = { point, line };  // 0=point, 1=line

  // ---- bump allocator over workspace ----
  const uintptr_t base = (uintptr_t)d_ws;
  size_t cur = 0;
  auto alloc = [&](size_t bytes) -> void* {
    cur = (cur + 255) & ~(size_t)255;
    void* p = (void*)(base + cur);
    cur += bytes;
    return p;
  };
  auto cdiv = [](size_t a, size_t b) { return (unsigned)((a + b - 1) / b); };

  const size_t SEQ  = (size_t)Bn * N0 * Cn;           // 3,276,800
  const size_t SEQ1 = (size_t)Bn * 1600 * Cn;
  const size_t SEQ2 = (size_t)Bn * 400 * Cn;
  const size_t NLv[3] = { (size_t)N0, 1600, 400 };

  // persistent per-branch buffers
  float *chw0[2], *seq0[2], *lvl1[2], *lvl2[2], *delta[2], *s1f[2], *finalf[2];
  float *embf[2][3];
  bf16  *qbf[2], *embb[2][3], *s1b[2];
  for (int br = 0; br < 2; ++br) {
    chw0[br]  = (float*)alloc(SEQ * 4);
    seq0[br]  = (float*)alloc(SEQ * 4);
    qbf[br]   = (bf16*)alloc(SEQ * 2);
    lvl1[br]  = (float*)alloc(SEQ1 * 4);
    lvl2[br]  = (float*)alloc(SEQ2 * 4);
    embf[br][0] = (float*)alloc(SEQ * 4);  embb[br][0] = (bf16*)alloc(SEQ * 2);
    embf[br][1] = (float*)alloc(SEQ1 * 4); embb[br][1] = (bf16*)alloc(SEQ1 * 2);
    embf[br][2] = (float*)alloc(SEQ2 * 4); embb[br][2] = (bf16*)alloc(SEQ2 * 2);
    delta[br] = (float*)alloc(SEQ * 4);
    s1f[br]   = (float*)alloc(SEQ * 4);
    s1b[br]   = (bf16*)alloc(SEQ * 2);
    finalf[br]= (float*)alloc(SEQ * 4);
  }
  // shared transients (reused sequentially); weight buffers padded to N%128==0
  float* chwT   = (float*)alloc(SEQ1 * 4);                       // level-1 NCHW
  bf16*  colb   = (bf16*)alloc((size_t)Bn * 1600 * K9 * 2);      // im2col
  bf16*  wkb    = (bf16*)alloc((size_t)K9 * Cn * 2);
  float* tmpseq = (float*)alloc(SEQ1 * 4);
  bf16*  woffb  = (bf16*)alloc((size_t)Cn * 256 * 2);   // 192 -> pad 256
  bf16*  wattnb = (bf16*)alloc((size_t)Cn * 128 * 2);   // 96  -> pad 128
  bf16*  wvalb  = (bf16*)alloc((size_t)Cn * Cn * 2);
  bf16*  woutb  = (bf16*)alloc((size_t)Cn * Cn * 2);
  bf16*  w1b    = (bf16*)alloc((size_t)Cn * 512 * 2);
  bf16*  w2b    = (bf16*)alloc((size_t)512 * Cn * 2);
  float* offf   = (float*)alloc((size_t)Bn * N0 * 192 * 4);
  float* awf    = (float*)alloc((size_t)Bn * N0 * 96 * 4);
  float* vf[3]  = { (float*)alloc(SEQ * 4), (float*)alloc(SEQ1 * 4),
                    (float*)alloc(SEQ2 * 4) };
  float* sampf  = (float*)alloc(SEQ * 4);
  bf16*  sampb  = (bf16*)alloc(SEQ * 2);
  bf16*  ffnh   = (bf16*)alloc((size_t)Bn * N0 * 512 * 2);
  float* ffno   = (float*)alloc(SEQ * 4);

  auto gemm = [&](const bf16* A, const bf16* Bw, const float* bias,
                  float* Cf, bf16* Cb, int M, int N, int K, int ldb, int act) {
    dim3 g((unsigned)(ldb / 128), (unsigned)((M + 63) / 64));
    k_gemm_bf16<<<g, 256, 0, stream>>>(A, Bw, bias, Cf, Cb, M, N, K, ldb, act);
  };
  auto cvtw = [&](bf16* dst, const float* src, int Kd, int Nd, int Npad) {
    k_cvt_pad<<<cdiv((size_t)Kd * Npad, 256), 256, 0, stream>>>(dst, src,
                                                                Kd, Nd, Npad);
  };
  auto cvt = [&](bf16* dst, const float* src, size_t n) {
    k_cvt_bf16<<<cdiv(n, 256), 256, 0, stream>>>(dst, src, n);
  };

  // ---- build both branches: pos-conv, queries, pyramid levels ----
  for (int br = 0; br < 2; ++br) {
    const BranchP& P = brp[br];
    k_dwconv_add<<<cdiv(SEQ, 256), 256, 0, stream>>>(chw0[br], seq0[br],
                                                     P.feats, P.pos);
    cvt(qbf[br], seq0[br], SEQ);
    k_add_embed<<<cdiv(SEQ, 256), 256, 0, stream>>>(
        embf[br][0], embb[br][0], seq0[br], P.embed + 0 * Cn, SEQ);
    // level 1: 80x80 -> 40x40 conv as GEMM (K=2304) + LN + embed
    k_conv_w_kn<<<cdiv((size_t)K9 * Cn, 256), 256, 0, stream>>>(wkb, P.conv0);
    k_im2col<<<cdiv((size_t)Bn * 1600 * K9, 256), 256, 0, stream>>>(
        colb, chw0[br], 80, 80, 40, 40);
    gemm(colb, wkb, nullptr, tmpseq, nullptr, Bn * 1600, Cn, K9, Cn, 0);
    k_layernorm<<<cdiv((size_t)Bn * 1600, 8), 256, 0, stream>>>(
        lvl1[br], (bf16*)nullptr, tmpseq, (const float*)nullptr,
        P.g0, P.ds_b0, Bn * 1600);
    k_add_embed<<<cdiv(SEQ1, 256), 256, 0, stream>>>(
        embf[br][1], embb[br][1], lvl1[br], P.embed + 1 * Cn, SEQ1);
    k_seq_to_chw<<<cdiv(SEQ1, 256), 256, 0, stream>>>(chwT, lvl1[br], 1600);
    // level 2: 40x40 -> 20x20
    k_conv_w_kn<<<cdiv((size_t)K9 * Cn, 256), 256, 0, stream>>>(wkb, P.conv1);
    k_im2col<<<cdiv((size_t)Bn * 400 * K9, 256), 256, 0, stream>>>(
        colb, chwT, 40, 40, 20, 20);
    gemm(colb, wkb, nullptr, tmpseq, nullptr, Bn * 400, Cn, K9, Cn, 0);
    k_layernorm<<<cdiv((size_t)Bn * 400, 8), 256, 0, stream>>>(
        lvl2[br], (bf16*)nullptr, tmpseq, (const float*)nullptr,
        P.g1, P.ds_b1, Bn * 400);
    k_add_embed<<<cdiv(SEQ2, 256), 256, 0, stream>>>(
        embf[br][2], embb[br][2], lvl2[br], P.embed + 2 * Cn, SEQ2);
  }

  // ---- cross deformable attention (queries qbr, values vbr) ----
  auto run_attn = [&](int qbr, int vbr, const AttnP& A, float* dOut) {
    cvtw(woffb,  A.W_off,  Cn, 192, 256);
    cvtw(wattnb, A.W_attn, Cn, 96, 128);
    cvtw(wvalb,  A.W_val,  Cn, Cn, Cn);
    cvtw(woutb,  A.W_out,  Cn, Cn, Cn);
    gemm(qbf[qbr], woffb,  A.b_off,  offf, nullptr, Bn * N0, 192, Cn, 256, 0);
    gemm(qbf[qbr], wattnb, A.b_attn, awf,  nullptr, Bn * N0, 96,  Cn, 128, 0);
    k_softmax24<<<cdiv((size_t)Bn * N0 * HEADS, 256), 256, 0, stream>>>(
        awf, Bn * N0 * HEADS);
    for (int l = 0; l < 3; ++l)
      gemm(embb[vbr][l], wvalb, A.b_val, vf[l], nullptr,
           (int)(Bn * NLv[l]), Cn, Cn, Cn, 0);
    k_sample<<<cdiv((size_t)Bn * N0 * HEADS * 32, 256), 256, 0, stream>>>(
        sampf, offf, awf, vf[0], vf[1], vf[2]);
    cvt(sampb, sampf, SEQ);
    gemm(sampb, woutb, A.b_out, dOut, nullptr, Bn * N0, Cn, Cn, Cn, 0);
  };
  run_attn(/*q=*/1, /*v=*/0, l2p, delta[1]);   // line queries -> point levels
  run_attn(/*q=*/0, /*v=*/1, p2l, delta[0]);   // point queries -> line levels

  // ---- residual + LN1, FFN (GELU), residual + LN2, output transpose ----
  float* outF = (float*)d_out;
  for (int br = 0; br < 2; ++br) {
    const BranchP& P = brp[br];
    const int rows = Bn * N0;
    k_layernorm<<<cdiv((size_t)rows, 8), 256, 0, stream>>>(
        s1f[br], s1b[br], seq0[br], delta[br], P.n1_g, P.n1_b, rows);
    cvtw(w1b, P.W1, Cn, 512, 512);
    cvtw(w2b, P.W2, 512, Cn, Cn);
    gemm(s1b[br], w1b, P.b1, nullptr, ffnh, rows, 512, Cn, 512, /*gelu*/1);
    gemm(ffnh,    w2b, P.b2, ffno, nullptr, rows, Cn, 512, Cn, 0);
    k_layernorm<<<cdiv((size_t)rows, 8), 256, 0, stream>>>(
        finalf[br], (bf16*)nullptr, s1f[br], ffno, P.n2_g, P.n2_b, rows);
    // output order: refined_point first, then refined_line
    k_seq_to_chw<<<cdiv(SEQ, 256), 256, 0, stream>>>(
        outF + (size_t)br * SEQ, finalf[br], N0);
  }
}
